// GNN_68736656605825
// MI455X (gfx1250) — matrix operand order repaired and checked by reading, not compiled, not run on hardware
//
#include <hip/hip_runtime.h>
#include <hip/hip_bf16.h>
#include <math.h>

#define N_NODES 50000
#define N_EDGES 800000
#define N_GRAPHS 64
#define NCLS 7
#define LEAKYF 0.2f
#define EPSF 1e-5f

typedef __attribute__((ext_vector_type(16))) _Float16 v16h;
typedef __attribute__((ext_vector_type(8)))  _Float16 v8h;
typedef __attribute__((ext_vector_type(8)))  float    v8f;

// ---- monotone float <-> uint mapping for atomicMax on floats ----
__device__ __forceinline__ unsigned ordF(float f) {
    unsigned u = __float_as_uint(f);
    return (u & 0x80000000u) ? ~u : (u | 0x80000000u);
}
__device__ __forceinline__ float unordF(unsigned u) {
    u = (u & 0x80000000u) ? (u ^ 0x80000000u) : ~u;
    return __uint_as_float(u);
}
// ordF(-inf) = ~0xFF800000
#define ORD_NEG_INF 0x007FFFFFu

// ---------------- fills ----------------
__global__ void fill_f32(float* p, float v, int n) {
    int i = blockIdx.x * 256 + threadIdx.x;
    if (i < n) p[i] = v;
}
__global__ void fill_u32(unsigned* p, unsigned v, int n) {
    int i = blockIdx.x * 256 + threadIdx.x;
    if (i < n) p[i] = v;
}

// =====================================================================
// Fast WMMA GEMM: C[M,Nd] = A[M,K] @ B[K,Nd]
// Requires M%16==0, K%64==0, Nd%64==0.  Block = 128 thr (4 waves),
// block tile = 16 rows x 64 cols, K stepped by 64 (2 WMMA per step).
// A staged row-major fp16; B staged TRANSPOSED ([n][k]) so each lane's
// fragment is 16 contiguous halves -> 2 x ds_load_b128.
// =====================================================================
__global__ void __launch_bounds__(128) wmma_gemm_fast(
    const float* __restrict__ A, const float* __restrict__ B,
    float* __restrict__ C, int M, int K, int Nd)
{
    __shared__ _Float16 Ash[16][64];   // [m][k], row stride 128B
    __shared__ _Float16 Bsh[64][72];   // [n][k] transposed, stride 144B (16B-aligned)
    const int t    = threadIdx.x;
    const int lane = t & 31;
    const int wave = t >> 5;
    const int m0 = blockIdx.y * 16;
    const int n0 = blockIdx.x * 64;
    const int half = lane >> 4;
    const int l16  = lane & 15;
    const int nw   = wave * 16;
    v8f acc = {};
    for (int k0 = 0; k0 < K; k0 += 64) {
        // A tile 16x64 f32: 1024 floats / 128 thr = 2 x float4 each
#pragma unroll
        for (int j = 0; j < 2; ++j) {
            int idx = t + j * 128;          // 16 float4 per row
            int r   = idx >> 4;
            int c4  = (idx & 15) << 2;
            float4 v = *(const float4*)(A + (size_t)(m0 + r) * K + k0 + c4);
            Ash[r][c4 + 0] = (_Float16)v.x;
            Ash[r][c4 + 1] = (_Float16)v.y;
            Ash[r][c4 + 2] = (_Float16)v.z;
            Ash[r][c4 + 3] = (_Float16)v.w;
        }
        // B tile 64x64 f32, stored transposed: 4096 floats / 128 thr = 8 x float4
#pragma unroll
        for (int j = 0; j < 8; ++j) {
            int idx = t + j * 128;
            int r   = idx >> 4;             // k
            int c4  = (idx & 15) << 2;      // n
            float4 v = *(const float4*)(B + (size_t)(k0 + r) * Nd + n0 + c4);
            Bsh[c4 + 0][r] = (_Float16)v.x;
            Bsh[c4 + 1][r] = (_Float16)v.y;
            Bsh[c4 + 2][r] = (_Float16)v.z;
            Bsh[c4 + 3][r] = (_Float16)v.w;
        }
        __syncthreads();
#pragma unroll
        for (int kk = 0; kk < 64; kk += 32) {
            // A fragment: K = kk + 16*half + i, contiguous 16 halves
            v8h alo = *(const v8h*)&Ash[l16][kk + 16 * half];
            v8h ahi = *(const v8h*)&Ash[l16][kk + 16 * half + 8];
            v16h a = __builtin_shufflevector(alo, ahi,
                0, 1, 2, 3, 4, 5, 6, 7, 8, 9, 10, 11, 12, 13, 14, 15);
            // B fragment: K = kk + 16*half + i, contiguous 16 halves
            v8h blo = *(const v8h*)&Bsh[nw + l16][kk + 16 * half];
            v8h bhi = *(const v8h*)&Bsh[nw + l16][kk + 16 * half + 8];
            v16h b = __builtin_shufflevector(blo, bhi,
                0, 1, 2, 3, 4, 5, 6, 7, 8, 9, 10, 11, 12, 13, 14, 15);
            acc = __builtin_amdgcn_wmma_f32_16x16x32_f16(
                false, a, false, b, (short)0, acc, false, false);
        }
        __syncthreads();
    }
#pragma unroll
    for (int r = 0; r < 8; ++r)
        C[(size_t)(m0 + r + 8 * half) * Nd + n0 + nw + l16] = acc[r];
}

// =====================================================================
// Generic guarded WMMA GEMM (+bias) for ragged shapes (K=5, Nd=7).
// =====================================================================
__global__ void __launch_bounds__(32) wmma_gemm_gen(
    const float* __restrict__ A, const float* __restrict__ B,
    const float* __restrict__ bias, float* __restrict__ C,
    int M, int K, int Nd)
{
    __shared__ float As[16][33];
    __shared__ float Bs[32][17];
    const int lane = threadIdx.x;
    const int n0 = blockIdx.x * 16;
    const int m0 = blockIdx.y * 16;
    const int half = lane >> 4;
    const int l16  = lane & 15;
    v8f acc = {};
    for (int k0 = 0; k0 < K; k0 += 32) {
#pragma unroll
        for (int u = 0; u < 16; ++u) {          // A tile 16x32
            int idx = lane + u * 32;
            int r = idx >> 5, c = idx & 31;
            int gm = m0 + r, gk = k0 + c;
            int cm = gm < M ? gm : M - 1;
            int ck = gk < K ? gk : K - 1;
            float v = A[(size_t)cm * K + ck];
            As[r][c] = (gm < M && gk < K) ? v : 0.f;
        }
#pragma unroll
        for (int u = 0; u < 16; ++u) {          // B tile 32x16
            int idx = lane + u * 32;
            int r = idx >> 4, c = idx & 15;
            int gk = k0 + r, gn = n0 + c;
            int ck = gk < K ? gk : K - 1;
            int cn = gn < Nd ? gn : Nd - 1;
            float v = B[(size_t)ck * Nd + cn];
            Bs[r][c] = (gk < K && gn < Nd) ? v : 0.f;
        }
        __syncthreads();
        v16h a, b;
#pragma unroll
        for (int i = 0; i < 16; ++i) {
            int ka = i + 16 * half;
            a[i] = (_Float16)As[l16][ka];
            int kb = i + 16 * half;
            b[i] = (_Float16)Bs[kb][l16];
        }
        acc = __builtin_amdgcn_wmma_f32_16x16x32_f16(
            false, a, false, b, (short)0, acc, false, false);
        __syncthreads();
    }
#pragma unroll
    for (int r = 0; r < 8; ++r) {
        int gm = m0 + r + 8 * half;
        int gn = n0 + l16;
        if (gm < M && gn < Nd) {
            float v = acc[r];
            if (bias) v += bias[gn];
            C[(size_t)gm * Nd + gn] = v;
        }
    }
}

// ---------------- per-node attention scores ----------------
template <int H, int C>
__global__ void att_scores(const float* __restrict__ xl,
                           const float* __restrict__ a_s, const float* __restrict__ a_d,
                           float* __restrict__ es, float* __restrict__ ed)
{
    int idx = blockIdx.x * blockDim.x + threadIdx.x;
    if (idx >= N_NODES * H) return;
    int n = idx / H, h = idx % H;
    const float* row = xl + (size_t)n * (H * C) + h * C;
    float s = 0.f, d = 0.f;
#pragma unroll 4
    for (int c = 0; c < C; c += 4) {
        float4 v  = *(const float4*)(row + c);
        float4 ws = *(const float4*)(a_s + h * C + c);
        float4 wd = *(const float4*)(a_d + h * C + c);
        s += v.x * ws.x + v.y * ws.y + v.z * ws.z + v.w * ws.w;
        d += v.x * wd.x + v.y * wd.y + v.z * wd.z + v.w * wd.w;
    }
    es[idx] = s;
    ed[idx] = d;
}

// ---------------- edge helpers ----------------
__device__ __forceinline__ void edge_sd(const int* __restrict__ ei, int e, int& s, int& d) {
    if (e < N_EDGES) { s = ei[e]; d = ei[N_EDGES + e]; }
    else { s = e - N_EDGES; d = s; }   // self-loops appended
}
template <int H>
__device__ __forceinline__ float edge_score(const float* es, const float* ed,
                                            int s, int d, int h) {
    float sc = es[s * H + h] + ed[d * H + h];
    return sc > 0.f ? sc : LEAKYF * sc;
}

template <int H>
__global__ void edge_max(const int* __restrict__ ei, const float* __restrict__ es,
                         const float* __restrict__ ed, unsigned* __restrict__ mU)
{
    int idx = blockIdx.x * blockDim.x + threadIdx.x;
    if (idx >= (N_EDGES + N_NODES) * H) return;
    int e = idx / H, h = idx % H;
    int s, d; edge_sd(ei, e, s, d);
    float sc = edge_score<H>(es, ed, s, d, h);
    atomicMax(&mU[d * H + h], ordF(sc));
}

template <int H>
__global__ void edge_expsum(const int* __restrict__ ei, const float* __restrict__ es,
                            const float* __restrict__ ed, const unsigned* __restrict__ mU,
                            float* __restrict__ den)
{
    int idx = blockIdx.x * blockDim.x + threadIdx.x;
    if (idx >= (N_EDGES + N_NODES) * H) return;
    int e = idx / H, h = idx % H;
    int s, d; edge_sd(ei, e, s, d);
    float sc = edge_score<H>(es, ed, s, d, h);
    float ex = expf(sc - unordF(mU[d * H + h]));
    atomicAdd(&den[d * H + h], ex);
}

template <int H, int C>
__global__ void edge_agg(const int* __restrict__ ei, const float* __restrict__ xl,
                         const float* __restrict__ es, const float* __restrict__ ed,
                         const unsigned* __restrict__ mU, const float* __restrict__ den,
                         float* __restrict__ out)
{
    int idx = blockIdx.x * blockDim.x + threadIdx.x;
    if (idx >= (N_EDGES + N_NODES) * H) return;
    int e = idx / H, h = idx % H;
    int s, d; edge_sd(ei, e, s, d);
    float sc = edge_score<H>(es, ed, s, d, h);
    float alpha = expf(sc - unordF(mU[d * H + h])) / den[d * H + h];
    const float* xr  = xl  + (size_t)s * (H * C) + h * C;
    float*       orr = out + (size_t)d * (H * C) + h * C;
    __builtin_prefetch(xr, 0, 1);   // global_prefetch_b8
#pragma unroll
    for (int c = 0; c < C; c += 4) {
        float4 v = *(const float4*)(xr + c);
        atomicAdd(&orr[c + 0], v.x * alpha);
        atomicAdd(&orr[c + 1], v.y * alpha);
        atomicAdd(&orr[c + 2], v.z * alpha);
        atomicAdd(&orr[c + 3], v.w * alpha);
    }
}

// ---------------- batch-norm (training stats) ----------------
__global__ void bn_stats(const float* __restrict__ x, const float* __restrict__ bias,
                         float* __restrict__ sum, float* __restrict__ sumsq,
                         int rows, int Cout, int rowsPerBlk)
{
    int ch = threadIdx.x;
    if (ch >= Cout) return;
    int r0 = blockIdx.x * rowsPerBlk;
    int r1 = r0 + rowsPerBlk; if (r1 > rows) r1 = rows;
    float b = bias ? bias[ch] : 0.f;
    float s = 0.f, s2 = 0.f;
    for (int r = r0; r < r1; ++r) {
        float v = x[(size_t)r * Cout + ch] + b;
        s += v; s2 += v * v;
    }
    atomicAdd(&sum[ch], s);
    atomicAdd(&sumsq[ch], s2);
}

// act: 0=none, 1=elu, 2=relu
__global__ void bn_apply(const float* __restrict__ x, const float* __restrict__ bias,
                         const float* __restrict__ sum, const float* __restrict__ sumsq,
                         const float* __restrict__ gam, const float* __restrict__ bet,
                         float* __restrict__ out, int rows, int Cout, int act)
{
    int idx = blockIdx.x * blockDim.x + threadIdx.x;
    if (idx >= rows * Cout) return;
    int ch = idx % Cout;
    float b = bias ? bias[ch] : 0.f;
    float inv = 1.f / (float)rows;
    float mu = sum[ch] * inv;
    float var = sumsq[ch] * inv - mu * mu;
    float y = (x[idx] + b - mu) * rsqrtf(var + EPSF) * gam[ch] + bet[ch];
    if (act == 1) y = y > 0.f ? y : expm1f(y);
    else if (act == 2) y = fmaxf(y, 0.f);
    out[idx] = y;
}

// ---------------- graph pooling ----------------
__global__ void pool_count(const int* __restrict__ batch, float* __restrict__ cnt) {
    int n = blockIdx.x * 256 + threadIdx.x;
    if (n < N_NODES) atomicAdd(&cnt[batch[n]], 1.f);
}
__global__ void pool_acc(const int* __restrict__ batch, const float* __restrict__ h,
                         float* __restrict__ zsum, unsigned* __restrict__ zmax)
{
    int idx = blockIdx.x * 256 + threadIdx.x;
    if (idx >= N_NODES * 128) return;
    int n = idx >> 7, ch = idx & 127;
    int b = batch[n];
    float v = h[(size_t)n * 128 + ch];
    atomicAdd(&zsum[b * 128 + ch], v);
    atomicMax(&zmax[b * 128 + ch], ordF(v));
}
__global__ void pool_fin(const float* __restrict__ cnt, const float* __restrict__ zsum,
                         const unsigned* __restrict__ zmax, float* __restrict__ z)
{
    int idx = blockIdx.x * 256 + threadIdx.x;
    if (idx >= N_GRAPHS * 128) return;
    int g = idx >> 7, ch = idx & 127;
    float c = cnt[g];
    z[g * 256 + ch]       = zsum[idx] / fmaxf(c, 1.f);
    z[g * 256 + 128 + ch] = (c > 0.f) ? unordF(zmax[idx]) : 0.f;
}

// ---------------- one GAT layer ----------------
template <int H, int C>
static void gat_layer(const float* hin, int Cin,
                      const float* W, const float* a_s, const float* a_d,
                      const float* bias, const float* gam, const float* bet,
                      float* xl, float* out,
                      float* es, float* ed, unsigned* mU, float* den,
                      float* sum, float* sumsq, const int* ei, hipStream_t stream)
{
    const int Cout = H * C;
    if ((Cin % 64) == 0 && (Cout % 64) == 0) {
        wmma_gemm_fast<<<dim3(Cout / 64, N_NODES / 16), 128, 0, stream>>>(
            hin, W, xl, N_NODES, Cin, Cout);
    } else {
        wmma_gemm_gen<<<dim3(Cout / 16, N_NODES / 16), 32, 0, stream>>>(
            hin, W, nullptr, xl, N_NODES, Cin, Cout);
    }

    const int nh = N_NODES * H;
    att_scores<H, C><<<(nh + 255) / 256, 256, 0, stream>>>(xl, a_s, a_d, es, ed);
    fill_u32<<<(nh + 255) / 256, 256, 0, stream>>>(mU, ORD_NEG_INF, nh);
    fill_f32<<<(nh + 255) / 256, 256, 0, stream>>>(den, 0.f, nh);

    const int tot = (N_EDGES + N_NODES) * H;
    edge_max<H><<<(tot + 255) / 256, 256, 0, stream>>>(ei, es, ed, mU);
    edge_expsum<H><<<(tot + 255) / 256, 256, 0, stream>>>(ei, es, ed, mU, den);

    const int ncout = N_NODES * Cout;
    fill_f32<<<(ncout + 255) / 256, 256, 0, stream>>>(out, 0.f, ncout);
    edge_agg<H, C><<<(tot + 255) / 256, 256, 0, stream>>>(ei, xl, es, ed, mU, den, out);

    fill_f32<<<1, 256, 0, stream>>>(sum, 0.f, Cout);
    fill_f32<<<1, 256, 0, stream>>>(sumsq, 0.f, Cout);
    const int RPB = 128;
    bn_stats<<<(N_NODES + RPB - 1) / RPB, Cout, 0, stream>>>(out, bias, sum, sumsq, N_NODES, Cout, RPB);
    bn_apply<<<(ncout + 255) / 256, 256, 0, stream>>>(out, bias, sum, sumsq, gam, bet, out, N_NODES, Cout, 1);
}

extern "C" void kernel_launch(void* const* d_in, const int* in_sizes, int n_in,
                              void* d_out, int out_size, void* d_ws, size_t ws_size,
                              hipStream_t stream)
{
    const float* x     = (const float*)d_in[0];
    const int*   ei    = (const int*)d_in[1];
    const int*   batch = (const int*)d_in[2];
    const float *W1 = (const float*)d_in[3],  *as1 = (const float*)d_in[4],  *ad1 = (const float*)d_in[5];
    const float *b1 = (const float*)d_in[6],  *g1  = (const float*)d_in[7],  *be1 = (const float*)d_in[8];
    const float *W2 = (const float*)d_in[9],  *as2 = (const float*)d_in[10], *ad2 = (const float*)d_in[11];
    const float *b2 = (const float*)d_in[12], *g2  = (const float*)d_in[13], *be2 = (const float*)d_in[14];
    const float *W3 = (const float*)d_in[15], *as3 = (const float*)d_in[16], *ad3 = (const float*)d_in[17];
    const float *b3 = (const float*)d_in[18], *g3  = (const float*)d_in[19], *be3 = (const float*)d_in[20];
    const float *Wf1 = (const float*)d_in[21], *bf1 = (const float*)d_in[22];
    const float *g4  = (const float*)d_in[23], *be4 = (const float*)d_in[24];
    const float *Wf2 = (const float*)d_in[25], *bf2 = (const float*)d_in[26];
    const float *g5  = (const float*)d_in[27], *be5 = (const float*)d_in[28];
    const float *Wf3 = (const float*)d_in[29], *bf3 = (const float*)d_in[30];

    float* wsf = (float*)d_ws;
    size_t o = 0;
    float*    P     = wsf + o; o += (size_t)N_NODES * 256;   // xl buffer
    float*    Q     = wsf + o; o += (size_t)N_NODES * 256;   // activation buffer
    float*    es    = wsf + o; o += (size_t)N_NODES * 4;
    float*    ed    = wsf + o; o += (size_t)N_NODES * 4;
    unsigned* mU    = (unsigned*)(wsf + o); o += (size_t)N_NODES * 4;
    float*    den   = wsf + o; o += (size_t)N_NODES * 4;
    float*    sum   = wsf + o; o += 256;
    float*    sumsq = wsf + o; o += 256;
    float*    cnt   = wsf + o; o += N_GRAPHS;
    float*    zsum  = wsf + o; o += N_GRAPHS * 128;
    unsigned* zmax  = (unsigned*)(wsf + o); o += N_GRAPHS * 128;
    float*    z     = wsf + o; o += N_GRAPHS * 256;
    float*    z2    = wsf + o; o += N_GRAPHS * 128;
    float*    z3    = wsf + o; o += N_GRAPHS * 64;
    (void)ws_size; (void)in_sizes; (void)n_in; (void)out_size;

    // GAT layers (ping-pong P=xl, Q=activations; hin dead after its GEMM)
    gat_layer<4, 64>(x, 5,    W1, as1, ad1, b1, g1, be1, P, Q, es, ed, mU, den, sum, sumsq, ei, stream);
    gat_layer<2, 128>(Q, 256, W2, as2, ad2, b2, g2, be2, P, Q, es, ed, mU, den, sum, sumsq, ei, stream);
    gat_layer<1, 128>(Q, 256, W3, as3, ad3, b3, g3, be3, P, Q, es, ed, mU, den, sum, sumsq, ei, stream);
    // Q now holds h3 [N,128]

    // pooling -> z [G, 256] = [mean | max]
    fill_f32<<<1, 256, 0, stream>>>(cnt, 0.f, N_GRAPHS);
    fill_f32<<<(N_GRAPHS * 128 + 255) / 256, 256, 0, stream>>>(zsum, 0.f, N_GRAPHS * 128);
    fill_u32<<<(N_GRAPHS * 128 + 255) / 256, 256, 0, stream>>>(zmax, ORD_NEG_INF, N_GRAPHS * 128);
    pool_count<<<(N_NODES + 255) / 256, 256, 0, stream>>>(batch, cnt);
    pool_acc<<<((size_t)N_NODES * 128 + 255) / 256, 256, 0, stream>>>(batch, Q, zsum, zmax);
    pool_fin<<<(N_GRAPHS * 128 + 255) / 256, 256, 0, stream>>>(cnt, zsum, zmax, z);

    // MLP head
    wmma_gemm_fast<<<dim3(128 / 64, N_GRAPHS / 16), 128, 0, stream>>>(z, Wf1, z2, N_GRAPHS, 256, 128);
    fill_f32<<<1, 256, 0, stream>>>(sum, 0.f, 128);
    fill_f32<<<1, 256, 0, stream>>>(sumsq, 0.f, 128);
    bn_stats<<<1, 128, 0, stream>>>(z2, bf1, sum, sumsq, N_GRAPHS, 128, N_GRAPHS);
    bn_apply<<<(N_GRAPHS * 128 + 255) / 256, 256, 0, stream>>>(z2, bf1, sum, sumsq, g4, be4, z2, N_GRAPHS, 128, 2);

    wmma_gemm_fast<<<dim3(64 / 64, N_GRAPHS / 16), 128, 0, stream>>>(z2, Wf2, z3, N_GRAPHS, 128, 64);
    fill_f32<<<1, 256, 0, stream>>>(sum, 0.f, 64);
    fill_f32<<<1, 256, 0, stream>>>(sumsq, 0.f, 64);
    bn_stats<<<1, 64, 0, stream>>>(z3, bf2, sum, sumsq, N_GRAPHS, 64, N_GRAPHS);
    bn_apply<<<(N_GRAPHS * 64 + 255) / 256, 256, 0, stream>>>(z3, bf2, sum, sumsq, g5, be5, z3, N_GRAPHS, 64, 2);

    wmma_gemm_gen<<<dim3((NCLS + 15) / 16, N_GRAPHS / 16), 32, 0, stream>>>(z3, Wf3, bf3, (float*)d_out, N_GRAPHS, 64, NCLS);
}
